// FRAPPLUS_63866163691796
// MI455X (gfx1250) — matrix operand order repaired and checked by reading, not compiled, not run on hardware
//
#include <hip/hip_runtime.h>

typedef __attribute__((ext_vector_type(16))) _Float16 v16h;
typedef __attribute__((ext_vector_type(8)))  _Float16 v8h;
typedef __attribute__((ext_vector_type(8)))  float    v8f;

#define NB    8            // batch elements per block
#define NPAIR 56           // P*(P-1)
#define ROWS  (NB * NPAIR) // 448 conv2 rows per block
#define NTILE (ROWS / 16)  // 28 WMMA M-tiles per block

__device__ __forceinline__ float sigmoidf(float x) {
    return 1.0f / (1.0f + __expf(-x));
}

// Sum across each DPP row of 16 lanes via row_ror 8/4/2/1 (pure VALU, no LDS).
__device__ __forceinline__ float row_sum16(float x) {
    int t;
    t = __builtin_amdgcn_update_dpp(0, __float_as_int(x), 0x128, 0xf, 0xf, true);
    x += __int_as_float(t);
    t = __builtin_amdgcn_update_dpp(0, __float_as_int(x), 0x124, 0xf, 0xf, true);
    x += __int_as_float(t);
    t = __builtin_amdgcn_update_dpp(0, __float_as_int(x), 0x122, 0xf, 0xf, true);
    x += __int_as_float(t);
    t = __builtin_amdgcn_update_dpp(0, __float_as_int(x), 0x121, 0xf, 0xf, true);
    x += __int_as_float(t);
    return x;
}

__global__ __launch_bounds__(128) void frap_wmma_kernel(
    const float* __restrict__ feat,
    const float* __restrict__ emb_phase,
    const float* __restrict__ Wv,  const float* __restrict__ bv,
    const float* __restrict__ Wl,  const float* __restrict__ bl,
    const float* __restrict__ emb_const,
    const float* __restrict__ Wcf, const float* __restrict__ bcf,
    const float* __restrict__ Wcc, const float* __restrict__ bcc,
    const float* __restrict__ Wcb, const float* __restrict__ bcb,
    const float* __restrict__ Wfin, const float* __restrict__ bfin,
    float* __restrict__ out, int Btot)
{
    __shared__ float sWl[128];
    __shared__ float sbl[16];
    __shared__ float sWv[4], sbv[4], sEmbP[8];
    __shared__ float sWcf[640];
    __shared__ float sbcf[20];
    __shared__ float sy[2][20];                 // relu(Wcc@emb_const[par]+bcc)
    __shared__ __align__(16) _Float16 sB[32][32];   // Wcb padded, [n=o'][k=o]
    __shared__ float sbcbE[32], sWfinE[32];     // zero-padded to 32
    __shared__ float spress[NB][8][16];         // pressure (phase 1a -> 1b)
    __shared__ float sApre[NB][8][20];          // Wcf[:, :16] @ pressure
    __shared__ float sBpre[NB][8][20];          // Wcf[:, 16:] @ pressure
    __shared__ __align__(16) _Float16 zin[ROWS][32]; // conv2 A-matrix rows (K padded)
    __shared__ float rowsum[ROWS];

    const int tid = threadIdx.x;

    // ---------------- Phase 0: stage weights / precompute constants --------
    if (tid < 128) sWl[tid] = Wl[tid];
    if (tid < 16)  sbl[tid] = bl[tid];
    if (tid < 4) { sWv[tid] = Wv[tid]; sbv[tid] = bv[tid]; }
    if (tid < 8)   sEmbP[tid] = emb_phase[tid];
    for (int i = tid; i < 640; i += 128) sWcf[i] = Wcf[i];
    if (tid < 20)  sbcf[tid] = bcf[tid];
    if (tid < 40) {                      // y depends only on (parity, channel)
        const int par = tid / 20, o = tid % 20;
        float acc = bcc[o];
        for (int c = 0; c < 4; ++c) acc += Wcc[o * 4 + c] * emb_const[par * 4 + c];
        sy[par][o] = fmaxf(acc, 0.0f);
    }
    for (int i = tid; i < 1024; i += 128) {   // B matrix: sB[n][k] = Wcb[n][k], padded
        const int n = i >> 5, k = i & 31;
        sB[n][k] = (n < 20 && k < 20) ? (_Float16)Wcb[n * 20 + k] : (_Float16)0.0f;
    }
    if (tid < 32) {
        sbcbE[tid]  = (tid < 20) ? bcb[tid]  : 0.0f;
        sWfinE[tid] = (tid < 20) ? Wfin[tid] : 0.0f;
    }
    // prefetch next block's feature rows (global_prefetch_b8)
    if (tid < NB) {
        const long pgb = (long)(blockIdx.x + 1) * NB + tid;
        if (pgb < (long)Btot) __builtin_prefetch(&feat[pgb * 16], 0, 0);
    }
    __syncthreads();

    // ---------------- Phase 1a: per-(batch, lane) pressure (fp32 VALU) -----
    if (tid < NB * 8) {
        const int bi = tid >> 3, l = tid & 7;
        long gb = (long)blockIdx.x * NB + bi;
        if (gb >= (long)Btot) gb = (long)Btot - 1;   // clamp (keeps loads in range)
        const float* f = feat + gb * 16;

        const int   pidx = (f[l] > 0.5f) ? 1 : 0;
        const float veh  = f[8 + l];

        float lane8[8];
        for (int d = 0; d < 4; ++d) {
            lane8[d]     = sigmoidf(veh * sWv[d] + sbv[d]);      // fv
            lane8[4 + d] = sigmoidf(sEmbP[pidx * 4 + d]);        // fp
        }
        for (int o = 0; o < 16; ++o) {
            float acc = sbl[o];
            for (int d = 0; d < 8; ++d) acc += lane8[d] * sWl[o * 8 + d];
            spress[bi][l][o] = 2.0f * fmaxf(acc, 0.0f);          // pressure = 2*relu
        }
    }
    __syncthreads();

    // ---------------- Phase 1b: Wcf projections, split 2 threads/(bi,l) ----
    {
        const int og  = tid & 1;        // half of the 20 output channels
        const int idx = tid >> 1;
        const int bi  = idx >> 3, l = idx & 7;
        float pr[16];
        for (int c = 0; c < 16; ++c) pr[c] = spress[bi][l][c];
        const int o0 = og * 10;
        for (int o = o0; o < o0 + 10; ++o) {
            float a = 0.0f, b2 = 0.0f;
            for (int c = 0; c < 16; ++c) {
                a  += sWcf[o * 32 + c]      * pr[c];
                b2 += sWcf[o * 32 + 16 + c] * pr[c];
            }
            sApre[bi][l][o] = a;
            sBpre[bi][l][o] = b2;
        }
    }
    __syncthreads();

    // ---------------- Phase 2: build conv2 input rows (f16, K padded to 32)
    for (int r = tid; r < ROWS; r += 128) {
        const int bi = r / NPAIR, p = r % NPAIR;
        const int i  = p / 7, jj = p % 7;
        const int j  = jj + ((jj >= i) ? 1 : 0);
        const int par = (i + j) & 1;
        v8h z0, z1, z2, z3;
        for (int o = 0; o < 8; ++o) {
            float x = fmaxf(sApre[bi][i][o] + sBpre[bi][j][o] + sbcf[o], 0.0f);
            z0[o] = (_Float16)(x * sy[par][o]);
        }
        for (int o = 8; o < 16; ++o) {
            float x = fmaxf(sApre[bi][i][o] + sBpre[bi][j][o] + sbcf[o], 0.0f);
            z1[o - 8] = (_Float16)(x * sy[par][o]);
        }
        for (int o = 16; o < 20; ++o) {
            float x = fmaxf(sApre[bi][i][o] + sBpre[bi][j][o] + sbcf[o], 0.0f);
            z2[o - 16] = (_Float16)(x * sy[par][o]);
        }
        for (int o = 4; o < 8; ++o) z2[o] = (_Float16)0.0f;
        for (int o = 0; o < 8; ++o) z3[o] = (_Float16)0.0f;
        *(v8h*)&zin[r][0]  = z0;     // 4x ds_store_b128 per row
        *(v8h*)&zin[r][8]  = z1;
        *(v8h*)&zin[r][16] = z2;
        *(v8h*)&zin[r][24] = z3;
    }
    __syncthreads();

    // ---------------- Phase 3: conv2 via v_wmma_f32_16x16x32_f16 ------------
    {
        const int wv = tid >> 5;         // wave id (4 waves)
        const int ln = tid & 31;
        const int n  = ln & 15;          // output column within N-tile
        const int hs = ln >> 4;          // half-wave select
        const int kb = hs * 16;          // B: K-range per half-wave

        // B fragments (constant across all tiles); layout: lane holds 16
        // contiguous K values of column n (halves 0..15 -> K kb..kb+15)
        v16h bf0, bf1;
        {
            v8h lo0 = *(const v8h*)&sB[n][kb];
            v8h hi0 = *(const v8h*)&sB[n][kb + 8];
            v8h lo1 = *(const v8h*)&sB[16 + n][kb];
            v8h hi1 = *(const v8h*)&sB[16 + n][kb + 8];
            for (int k = 0; k < 8; ++k) {
                bf0[k] = lo0[k]; bf0[8 + k] = hi0[k];
                bf1[k] = lo1[k]; bf1[8 + k] = hi1[k];
            }
        }
        const float bc0 = sbcbE[n],      wf0 = sWfinE[n];
        const float bc1 = sbcbE[16 + n], wf1 = sWfinE[16 + n];

        // 28 tiles over 4 waves -> uniform 7 iterations (EXEC stays all-ones)
        for (int tile = wv; tile < NTILE; tile += 4) {
            // A fragment: contiguous K order (same as B) — lane<16: K {0..15},
            // lane>=16: K {16..31}, row M = tile*16 + (lane&15)
            const int arow = tile * 16 + n;
            const int aoff = hs * 16;
            v8h alo = *(const v8h*)&zin[arow][aoff];
            v8h ahi = *(const v8h*)&zin[arow][aoff + 8];
            v16h af;
            for (int k = 0; k < 8; ++k) { af[k] = alo[k]; af[8 + k] = ahi[k]; }

            v8f c0 = {0.f,0.f,0.f,0.f,0.f,0.f,0.f,0.f};
            v8f c1 = {0.f,0.f,0.f,0.f,0.f,0.f,0.f,0.f};
            c0 = __builtin_amdgcn_wmma_f32_16x16x32_f16(false, af, false, bf0,
                                                        (short)0, c0, false, false);
            c1 = __builtin_amdgcn_wmma_f32_16x16x32_f16(false, af, false, bf1,
                                                        (short)0, c1, false, false);

            // fused epilogue: relu(+bcb) * Wfin, then reduce the 16 output
            // columns per half-row with DPP row_ror adds (D layout: lanes
            // 0-15 = rows 0-7, lanes 16-31 = rows 8-15)
            float s[8];
            for (int r = 0; r < 8; ++r) {
                float v = fmaxf(c0[r] + bc0, 0.0f) * wf0 +
                          fmaxf(c1[r] + bc1, 0.0f) * wf1;
                s[r] = row_sum16(v);
            }
            if (n == 0) {
                const int mb = hs * 8;
                for (int r = 0; r < 8; ++r)
                    rowsum[tile * 16 + mb + r] = s[r];
            }
        }
    }
    __syncthreads();

    // ---------------- Phase 4: sum over 7 j's per (b,i), add 7*bfin --------
    if (tid < NB * 8) {
        const int bi = tid >> 3, i = tid & 7;
        const long gb = (long)blockIdx.x * NB + bi;
        if (gb < (long)Btot) {
            float acc = 7.0f * bfin[0];
            for (int jj = 0; jj < 7; ++jj)
                acc += rowsum[bi * NPAIR + i * 7 + jj];
            out[gb * 8 + i] = acc;
        }
    }
}

extern "C" void kernel_launch(void* const* d_in, const int* in_sizes, int n_in,
                              void* d_out, int out_size, void* d_ws, size_t ws_size,
                              hipStream_t stream) {
    const float* feat      = (const float*)d_in[0];
    const float* emb_phase = (const float*)d_in[1];
    const float* Wv        = (const float*)d_in[2];
    const float* bv        = (const float*)d_in[3];
    const float* Wl        = (const float*)d_in[4];
    const float* bl        = (const float*)d_in[5];
    const float* emb_const = (const float*)d_in[6];
    const float* Wcf       = (const float*)d_in[7];
    const float* bcf       = (const float*)d_in[8];
    const float* Wcc       = (const float*)d_in[9];
    const float* bcc       = (const float*)d_in[10];
    const float* Wcb       = (const float*)d_in[11];
    const float* bcb       = (const float*)d_in[12];
    const float* Wfin      = (const float*)d_in[13];
    const float* bfin      = (const float*)d_in[14];
    float* out = (float*)d_out;

    const int Btot   = in_sizes[0] / 16;
    const int blocks = (Btot + NB - 1) / NB;
    frap_wmma_kernel<<<blocks, 128, 0, stream>>>(
        feat, emb_phase, Wv, bv, Wl, bl, emb_const, Wcf, bcf,
        Wcc, bcc, Wcb, bcb, Wfin, bfin, out, Btot);
}
